// CriticGCN_36859409334422
// MI455X (gfx1250) — matrix operand order repaired and checked by reading, not compiled, not run on hardware
//
#include <hip/hip_runtime.h>
#include <hip/hip_bf16.h>

typedef __attribute__((ext_vector_type(16))) _Float16 v16h;
typedef __attribute__((ext_vector_type(8)))  float    v8f;

// ---------------------------------------------------------------------------
// WMMA wrapper: D = A(16x32 f16) * B(32x16 f16) + C(16x16 f32)
// ---------------------------------------------------------------------------
__device__ __forceinline__ v8f wmma_f16(v16h a, v16h b, v8f c) {
  return __builtin_amdgcn_wmma_f32_16x16x32_f16(
      /*neg_a=*/false, a, /*neg_b=*/false, b,
      /*c_mod=*/(short)0, c, /*reuse_a=*/false, /*reuse_b=*/false);
}

__device__ __forceinline__ unsigned int pack2h(float f0, float f1) {
  union { unsigned int u; _Float16 h[2]; } p;
  p.h[0] = (_Float16)f0;
  p.h[1] = (_Float16)f1;
  return p.u;
}

// A-fragment K index for (lane, pair i): lanes 0-15 hold K {0..7,16..23},
// lanes 16-31 hold K {8..15,24..31} (ISA 7.12.2, 16-bit A 16x32)
__device__ __forceinline__ int a_k0(int i, int lane) {
  return (i < 4 ? 2 * i : 16 + 2 * (i - 4)) + 8 * (lane >> 4);
}

// ---------------------------------------------------------------------------
// degree / normalization
// ---------------------------------------------------------------------------
__global__ void k_zero_f32(float* p, int n) {
  int g = blockIdx.x * blockDim.x + threadIdx.x;
  if (g < n) p[g] = 0.0f;
}

__global__ void k_deg_accum(const int* __restrict__ dstI, float* __restrict__ deg, int E) {
  int g = blockIdx.x * blockDim.x + threadIdx.x;
  if (g < E) atomicAdd(&deg[dstI[g]], 1.0f);
}

__global__ void k_dis_fin(float* d, int n) {
  int g = blockIdx.x * blockDim.x + threadIdx.x;
  if (g < n) d[g] = rsqrtf(d[g] + 1.0f);  // +1 = self loop; deg+1 > 0 always
}

// ---------------------------------------------------------------------------
// Pack node features (concat state|edge_attr, pad F->Kpad) into A fragments.
// dword g -> frag = g>>8 (= mt*Ktiles+kt), lane = (g>>3)&31, i = g&7
// ---------------------------------------------------------------------------
__global__ void k_pack_x(const float* __restrict__ state, const float* __restrict__ eattr,
                         int N1, int N, int F, int Ktiles,
                         unsigned int* __restrict__ dst, int total) {
  int g = blockIdx.x * blockDim.x + threadIdx.x;
  if (g >= total) return;
  int i = g & 7, lane = (g >> 3) & 31, frag = g >> 8;
  int kt = frag % Ktiles, mt = frag / Ktiles;
  int m  = mt * 16 + (lane & 15);
  int k0 = kt * 32 + a_k0(i, lane);
  float f0 = 0.0f, f1 = 0.0f;
  if (m < N) {
    const float* row = (m < N1) ? (state + (size_t)m * F) : (eattr + (size_t)(m - N1) * F);
    if (k0 < F)     f0 = row[k0];
    if (k0 + 1 < F) f1 = row[k0 + 1];
  }
  dst[g] = pack2h(f0, f1);
}

// ---------------------------------------------------------------------------
// Pack weight matrix W[Kreal x Nreal] row-major f32 into B fragments.
// B 32x16 f16: lane n = lane&15, K = kt*32 + a_k0(i, lane) (same K order as A)
// frag index = kt*Ntiles + nt
// ---------------------------------------------------------------------------
__global__ void k_pack_w(const float* __restrict__ W, int Kreal, int Nreal, int Ntiles,
                         unsigned int* __restrict__ dst, int total) {
  int g = blockIdx.x * blockDim.x + threadIdx.x;
  if (g >= total) return;
  int i = g & 7, lane = (g >> 3) & 31, frag = g >> 8;
  int nt = frag % Ntiles, kt = frag / Ntiles;
  int k0 = kt * 32 + a_k0(i, lane);
  int n  = nt * 16 + (lane & 15);
  float f0 = 0.0f, f1 = 0.0f;
  if (n < Nreal) {
    if (k0 < Kreal)     f0 = W[(size_t)k0 * Nreal + n];
    if (k0 + 1 < Kreal) f1 = W[(size_t)(k0 + 1) * Nreal + n];
  }
  dst[g] = pack2h(f0, f1);
}

// ---------------------------------------------------------------------------
// GEMM: one wave computes a 16x64 strip (4 N-tiles reuse one A fragment).
// Ap frags: [mt*Ktiles+kt], Bp frags: [kt*Ntiles+nt]; frag = 32 v16h chunks.
// Output row-major f16 with pitch P.
// ---------------------------------------------------------------------------
__global__ void k_gemm_wmma(const v16h* __restrict__ Ap, const v16h* __restrict__ Bp,
                            _Float16* __restrict__ C,
                            int Mtiles, int Ktiles, int Ntiles, int P) {
  const int wave = blockIdx.x * (blockDim.x >> 5) + (threadIdx.x >> 5);
  const int lane = threadIdx.x & 31;
  const int ng = Ntiles >> 2;
  if (wave >= Mtiles * ng) return;
  const int mt  = wave / ng;
  const int nt0 = (wave % ng) << 2;

  v8f acc0 = {}; v8f acc1 = {}; v8f acc2 = {}; v8f acc3 = {};
  for (int kt = 0; kt < Ktiles; ++kt) {
    v16h af = Ap[(size_t)(mt * Ktiles + kt) * 32 + lane];
    v16h b0 = Bp[(size_t)(kt * Ntiles + nt0 + 0) * 32 + lane];
    acc0 = wmma_f16(af, b0, acc0);
    v16h b1 = Bp[(size_t)(kt * Ntiles + nt0 + 1) * 32 + lane];
    acc1 = wmma_f16(af, b1, acc1);
    v16h b2 = Bp[(size_t)(kt * Ntiles + nt0 + 2) * 32 + lane];
    acc2 = wmma_f16(af, b2, acc2);
    v16h b3 = Bp[(size_t)(kt * Ntiles + nt0 + 3) * 32 + lane];
    acc3 = wmma_f16(af, b3, acc3);
  }

  const int ncol = nt0 * 16 + (lane & 15);
  const int mrow = mt * 16 + ((lane >> 4) << 3);  // C/D: rows m = i + 8*(lane>=16)
  _Float16* cp = C + (size_t)mrow * P + ncol;
#pragma unroll
  for (int i = 0; i < 8; ++i) {
    size_t ro = (size_t)i * P;
    cp[ro +  0] = (_Float16)acc0[i];
    cp[ro + 16] = (_Float16)acc1[i];
    cp[ro + 32] = (_Float16)acc2[i];
    cp[ro + 48] = (_Float16)acc3[i];
  }
}

// ---------------------------------------------------------------------------
// Aggregation init: Acc[r,c] = dis[r]^2 * H[r,c]   (self-loop term; also
// serves as the buffer initialization). Rows >= N (padding) -> 0.
// ---------------------------------------------------------------------------
__global__ void k_agg_init(const _Float16* __restrict__ H, const float* __restrict__ dis,
                           float* __restrict__ Acc, int N, int P, int total) {
  int g = blockIdx.x * blockDim.x + threadIdx.x;
  if (g >= total) return;
  int r = g / P;
  float w = (r < N) ? dis[r] : 0.0f;
  Acc[g] = w * w * (float)H[g];
}

// ---------------------------------------------------------------------------
// Edge scatter: Acc[d, :] += dis[s]*dis[d] * H[s, :] ; 4 columns per thread.
// ---------------------------------------------------------------------------
__global__ void k_agg_edges(const int* __restrict__ srcI, const int* __restrict__ dstI,
                            const _Float16* __restrict__ H, const float* __restrict__ dis,
                            float* __restrict__ Acc, int E, int P) {
  int cpt = P >> 2;  // column-quads per edge
  int g = blockIdx.x * blockDim.x + threadIdx.x;
  if (g >= E * cpt) return;
  int e  = g / cpt;
  int c0 = (g - e * cpt) << 2;
  int s = srcI[e], d = dstI[e];
  float w = dis[s] * dis[d];
  const _Float16* hp = H + (size_t)s * P + c0;
  float* op = Acc + (size_t)d * P + c0;
  atomicAdd(op + 0, w * (float)hp[0]);
  atomicAdd(op + 1, w * (float)hp[1]);
  atomicAdd(op + 2, w * (float)hp[2]);
  atomicAdd(op + 3, w * (float)hp[3]);
}

// ---------------------------------------------------------------------------
// bias + ReLU + repack Acc(f32, pitch P) into A fragments (f16) for next GEMM
// ---------------------------------------------------------------------------
__global__ void k_cvt_pack(const float* __restrict__ Acc, const float* __restrict__ bias,
                           int Hreal, int P, int Ktiles,
                           unsigned int* __restrict__ dst, int total) {
  int g = blockIdx.x * blockDim.x + threadIdx.x;
  if (g >= total) return;
  int i = g & 7, lane = (g >> 3) & 31, frag = g >> 8;
  int kt = frag % Ktiles, mt = frag / Ktiles;
  int m  = mt * 16 + (lane & 15);
  int k0 = kt * 32 + a_k0(i, lane);
  float f0 = Acc[(size_t)m * P + k0];
  float f1 = Acc[(size_t)m * P + k0 + 1];
  if (k0 < Hreal)     f0 += bias[k0];
  if (k0 + 1 < Hreal) f1 += bias[k0 + 1];
  f0 = fmaxf(f0, 0.0f);
  f1 = fmaxf(f1, 0.0f);
  dst[g] = pack2h(f0, f1);
}

// ---------------------------------------------------------------------------
// Final: out[r] = b3 + sum_c (Acc[r,c] + b2[c]) * W3[c]; one wave per row.
// ---------------------------------------------------------------------------
__global__ void k_final(const float* __restrict__ Acc, const float* __restrict__ b2,
                        const float* __restrict__ W3, const float* __restrict__ b3,
                        float* __restrict__ out, int N, int Hreal, int P) {
  int wave = blockIdx.x * (blockDim.x >> 5) + (threadIdx.x >> 5);
  int lane = threadIdx.x & 31;
  if (wave >= N) return;
  const float* row = Acc + (size_t)wave * P;
  float acc = 0.0f;
  for (int c = lane; c < Hreal; c += 32)
    acc += (row[c] + b2[c]) * W3[c];
#pragma unroll
  for (int off = 16; off > 0; off >>= 1)
    acc += __shfl_down(acc, off, 32);
  if (lane == 0) out[wave] = acc + b3[0];
}

// ---------------------------------------------------------------------------
extern "C" void kernel_launch(void* const* d_in, const int* in_sizes, int n_in,
                              void* d_out, int out_size, void* d_ws, size_t ws_size,
                              hipStream_t stream) {
  const float* state = (const float*)d_in[0];
  const float* eattr = (const float*)d_in[1];
  const int*   eidx  = (const int*)d_in[2];
  const float* W1 = (const float*)d_in[3];
  const float* b1 = (const float*)d_in[4];
  const float* W2 = (const float*)d_in[5];
  const float* b2 = (const float*)d_in[6];
  const float* W3 = (const float*)d_in[7];
  const float* b3 = (const float*)d_in[8];
  float* out = (float*)d_out;

  const int Hd = in_sizes[4];             // 500
  const int F  = in_sizes[3] / Hd;        // 20
  const int N1 = in_sizes[0] / F;         // 10000
  const int N  = N1 + in_sizes[1] / F;    // 50000
  const int E  = in_sizes[2] / 2;         // 800000
  const int* srcI = eidx;
  const int* dstI = eidx + E;

  const int Mtiles = (N + 15) / 16;            // 3125 (exact: 50000 = 16*3125)
  const int Nr     = Mtiles * 16;
  const int P      = ((Hd + 31) / 32) * 32;    // 512 (pitch; mult of 32 for K)
  const int Ntiles = P / 16;                   // 32
  const int Kt1    = ((F + 31) / 32);          // 1
  const int Kt2    = P / 32;                   // 16

  // ---- workspace carve-up (256B aligned) ----
  char* ws = (char*)d_ws;
  size_t off = 0;
  auto carve = [&](size_t bytes) -> char* {
    char* p = ws + off;
    off = (off + bytes + 255) & ~(size_t)255;
    return p;
  };
  float*        dis  = (float*)carve((size_t)N * 4);
  unsigned int* Xp   = (unsigned int*)carve((size_t)Mtiles * Kt1 * 256 * 4);
  unsigned int* W1p  = (unsigned int*)carve((size_t)Kt1 * Ntiles * 256 * 4);
  unsigned int* W2p  = (unsigned int*)carve((size_t)Kt2 * Ntiles * 256 * 4);
  _Float16*     Hh   = (_Float16*)carve((size_t)Nr * P * 2);
  float*        Acc  = (float*)carve((size_t)Nr * P * 4);
  unsigned int* A2p  = (unsigned int*)carve((size_t)Mtiles * Kt2 * 256 * 4);
  (void)ws_size;

  const int TB = 256;
  auto NB = [](long long n) { return (unsigned)((n + 255) / 256); };

  // ---- degree / normalization ----
  k_zero_f32<<<NB(N), TB, 0, stream>>>(dis, N);
  k_deg_accum<<<NB(E), TB, 0, stream>>>(dstI, dis, E);
  k_dis_fin<<<NB(N), TB, 0, stream>>>(dis, N);

  // ---- pack operands ----
  int totX = Mtiles * Kt1 * 256;
  k_pack_x<<<NB(totX), TB, 0, stream>>>(state, eattr, N1, N, F, Kt1, Xp, totX);
  int totW1 = Kt1 * Ntiles * 256;
  k_pack_w<<<NB(totW1), TB, 0, stream>>>(W1, F, Hd, Ntiles, W1p, totW1);
  int totW2 = Kt2 * Ntiles * 256;
  k_pack_w<<<NB(totW2), TB, 0, stream>>>(W2, Hd, Hd, Ntiles, W2p, totW2);

  const int waves1 = Mtiles * (Ntiles >> 2);
  const int totNP  = Nr * P;
  const int eThreads = E * (P >> 2);

  // ---- layer 1: H = X@W1 ; Acc = scatter ; A2p = relu(Acc+b1) packed ----
  k_gemm_wmma<<<NB((long long)waves1 * 32), TB, 0, stream>>>(
      (const v16h*)Xp, (const v16h*)W1p, Hh, Mtiles, Kt1, Ntiles, P);
  k_agg_init<<<NB(totNP), TB, 0, stream>>>(Hh, dis, Acc, N, P, totNP);
  k_agg_edges<<<NB(eThreads), TB, 0, stream>>>(srcI, dstI, Hh, dis, Acc, E, P);
  int totA2 = Mtiles * Kt2 * 256;
  k_cvt_pack<<<NB(totA2), TB, 0, stream>>>(Acc, b1, Hd, P, Kt2, A2p, totA2);

  // ---- layer 2: H = A@W2 ; Acc = scatter ----
  k_gemm_wmma<<<NB((long long)waves1 * 32), TB, 0, stream>>>(
      (const v16h*)A2p, (const v16h*)W2p, Hh, Mtiles, Kt2, Ntiles, P);
  k_agg_init<<<NB(totNP), TB, 0, stream>>>(Hh, dis, Acc, N, P, totNP);
  k_agg_edges<<<NB(eThreads), TB, 0, stream>>>(srcI, dstI, Hh, dis, Acc, E, P);

  // ---- output head ----
  k_final<<<NB((long long)N * 32), TB, 0, stream>>>(Acc, b2, W3, b3, out, N, Hd, P);
  (void)out_size; (void)n_in;
}
